// DenoiserNetwork_88295937671432
// MI455X (gfx1250) — matrix operand order repaired and checked by reading, not compiled, not run on hardware
//
#include <hip/hip_runtime.h>
#include <hip/hip_bf16.h>

// CDNA5 / gfx1250: wave32, v_wmma_f32_16x16x32_bf16.
typedef __attribute__((ext_vector_type(16))) __bf16 v16bf;
typedef __attribute__((ext_vector_type(8)))  __bf16 v8bf;
typedef __attribute__((ext_vector_type(4)))  __bf16 v4bf;
typedef __attribute__((ext_vector_type(8)))  float  v8f;

// ---------------------------------------------------------------------------
// Prep: RepVGG reparameterization. Fuse w5 + pad(w3) + pad(w1) (+ identity)
// into one 5x5 kernel, packed as a 16-bit WMMA A fragment in contiguous K order:
//   lane M = lane&15; lanes 0-15: halves 0..15 -> K 0..15
//                     lanes 16-31: halves 0..15 -> K 16..31.
// GEMM K ordering: k = tap*Cpad + c  (tap = ky*5+kx, Cpad = padded in-chans).
// ---------------------------------------------------------------------------
__global__ __launch_bounds__(256) void prep_layer_kernel(
    const float* __restrict__ w5, const float* __restrict__ b5,
    const float* __restrict__ w3, const float* __restrict__ b3,
    const float* __restrict__ w1, const float* __restrict__ b1,
    __bf16* __restrict__ aFrag, float* __restrict__ bias,
    int Ci, int Cpad, int Co, int addIdentity, int kIters)
{
    const int tid = threadIdx.x;
    const int total = kIters * 512;               // kIters * 32 lanes * 16 halves
    for (int idx = tid; idx < total; idx += 256) {
        int kIter = idx >> 9;
        int rem   = idx & 511;
        int lane  = rem >> 4;
        int h     = rem & 15;
        int kL = ((lane & 16) ? 16 : 0) + h;
        int k  = kIter * 32 + kL;
        int m  = lane & 15;                       // output channel
        int tap = k / Cpad;
        int c   = k - tap * Cpad;
        float v = 0.0f;
        if (m < Co && tap < 25 && c < Ci) {
            int ky = tap / 5;
            int kx = tap - ky * 5;
            v = w5[((m * Ci + c) * 5 + ky) * 5 + kx];
            if (ky >= 1 && ky <= 3 && kx >= 1 && kx <= 3)
                v += w3[((m * Ci + c) * 3 + (ky - 1)) * 3 + (kx - 1)];
            if (ky == 2 && kx == 2) {
                v += w1[m * Ci + c];
                if (addIdentity && c == m) v += 1.0f;   // residual folded in
            }
        }
        aFrag[idx] = (__bf16)v;                  // idx == (kIter*32+lane)*16 + h
    }
    if (tid < 16) bias[tid] = (tid < Co) ? (b5[tid] + b3[tid] + b1[tid]) : 0.0f;
}

// ---------------------------------------------------------------------------
// Layers 1..3: Ci=16 NHWC bf16 input. Block = 16x16 pixel tile, 8 wave32s;
// each wave owns two adjacent rows -> two independent WMMA accumulator
// chains sharing one A fragment (A traffic halved, XDL latency hidden).
// B layout (32x16 bf16): lanes 0-15 hold K 0..15, lanes 16-31 hold K 16..31,
// K sequential across halves -> one lane = one full tap (16 channels) = one
// 32B contiguous LDS pixel slot -> 2x ds_load_b128, offsets compile-time.
// ---------------------------------------------------------------------------
template <int Co, bool DST_F32>
__global__ __launch_bounds__(256) void conv16_wmma(
    const __bf16* __restrict__ src, void* __restrict__ dst,
    const __bf16* __restrict__ aFrag, const float* __restrict__ bias)
{
    constexpr int KITERS = 13;                 // K = 25*16 = 400 -> 416
    constexpr int LDIM   = 20;                 // 16 + 2 halo each side
    __shared__ v16bf lds[LDIM * LDIM + 1];     // 400 halo pixels NHWC16 + zero slot

    const int img     = blockIdx.z;
    const int tileRow = blockIdx.y;            // * 16 rows
    const int tileCol = blockIdx.x;            // * 16 cols
    const int tid     = threadIdx.x;

    // stage 20x20 halo tile, one 32B NHWC pixel per slot (2x b128 in/out)
    for (int p = tid; p < LDIM * LDIM; p += 256) {
        int row = p / LDIM;
        int col = p - row * LDIM;
        int gr = tileRow * 16 + row - 2;
        int gc = tileCol * 16 + col - 2;
        v16bf v = {};
        if (gr >= 0 && gr < 1024 && gc >= 0 && gc < 1024) {
            size_t pix = (((size_t)img << 10) + gr);
            pix = (pix << 10) + gc;
            v = *(const v16bf*)(src + pix * 16);
        }
        lds[p] = v;
    }
    if (tid == 0) { v16bf z = {}; lds[LDIM * LDIM] = z; }   // zero slot
    __syncthreads();

    const int lane = tid & 31;
    const int wv   = tid >> 5;                 // wave id -> rows 2*wv, 2*wv+1
    const int n    = lane & 15;                // pixel column (B/D column)
    const int hi   = lane >> 4;                // K-half / M-half select
    const int pb0  = (2 * wv) * LDIM + n;      // LDS slot of tap (0,0), row 0
    const int pb1  = pb0 + LDIM;               // row 1

    v8f acc0 = {}, acc1 = {};
#pragma unroll
    for (int kIter = 0; kIter < KITERS; ++kIter) {
        const v16bf a = *(const v16bf*)(aFrag + ((kIter * 32 + lane) << 4));
        const int t0 = 2 * kIter;              // compile-time taps
        const int t1 = 2 * kIter + 1;
        const int o0 = (t0 / 5) * LDIM + (t0 % 5);
        const int o1 = (t1 < 25) ? ((t1 / 5) * LDIM + (t1 % 5)) : 0;
        const bool pad = (t1 >= 25);
        const int ia = pb0 + o0;
        const int ib = pad ? (LDIM * LDIM) : (pb0 + o1);
        const int ic = pb1 + o0;
        const int id = pad ? (LDIM * LDIM) : (pb1 + o1);
        const v16bf b0 = lds[hi ? ib : ia];
        const v16bf b1 = lds[hi ? id : ic];
        acc0 = __builtin_amdgcn_wmma_f32_16x16x32_bf16(
            false, a, false, b0, (short)0, acc0, false, false);
        acc1 = __builtin_amdgcn_wmma_f32_16x16x32_bf16(
            false, a, false, b1, (short)0, acc1, false, false);
    }

    // epilogue: lane holds channels hi*8..hi*8+7 of two pixels; packed stores
    const int gc  = tileCol * 16 + n;
    const int gr0 = tileRow * 16 + 2 * wv;
    size_t pix0 = (((size_t)img << 10) + gr0);
    pix0 = (pix0 << 10) + gc;
    size_t pix1 = pix0 + 1024;                 // next row
    if (DST_F32) {
        if (hi == 0) {                         // Co == 8: only channels 0..7
            float* g0 = (float*)dst + pix0 * 8;
            float* g1 = (float*)dst + pix1 * 8;
#pragma unroll
            for (int r = 0; r < 8; ++r) {
                float v0 = acc0[r] + bias[r];
                float v1 = acc1[r] + bias[r];
                g0[r] = v0 > 0.0f ? v0 : 0.0f;
                g1[r] = v1 > 0.0f ? v1 : 0.0f;
            }
        }
    } else {                                   // Co == 16, NHWC bf16
        v8bf o0, o1;
#pragma unroll
        for (int r = 0; r < 8; ++r) {
            float b = bias[hi * 8 + r];
            float v0 = acc0[r] + b;
            float v1 = acc1[r] + b;
            o0[r] = (__bf16)(v0 > 0.0f ? v0 : 0.0f);
            o1[r] = (__bf16)(v1 > 0.0f ? v1 : 0.0f);
        }
        *(v8bf*)((__bf16*)dst + pix0 * 16 + hi * 8) = o0;   // b128 stores
        *(v8bf*)((__bf16*)dst + pix1 * 16 + hi * 8) = o1;
    }
}

// ---------------------------------------------------------------------------
// Layer 0: Ci=3 padded to 4 (K = 25*4 = 100 -> 128, 4 WMMAs). Input is
// channel-planar f32; staged as v4bf (8B) NHWC pixels. Both candidate taps
// per (kIter,j) are compile-time -> B built from 4x (cndmask + ds_load_b64).
// ---------------------------------------------------------------------------
__global__ __launch_bounds__(256) void conv_in_wmma(
    const float* __restrict__ src, __bf16* __restrict__ dst,
    const __bf16* __restrict__ aFrag, const float* __restrict__ bias)
{
    constexpr int KITERS = 4;
    __shared__ v4bf lds4[241];                 // 240 halo pixels NHWC4 + zero slot

    const int img     = blockIdx.z;
    const int tileRow = blockIdx.y;            // * 8 rows
    const int tileCol = blockIdx.x;            // * 16 cols
    const int tid     = threadIdx.x;

    for (int p = tid; p < 240; p += 256) {
        int row = p / 20;
        int col = p - row * 20;
        int gr = tileRow * 8 + row - 2;
        int gc = tileCol * 16 + col - 2;
        v4bf v = {};
        if (gr >= 0 && gr < 1024 && gc >= 0 && gc < 1024) {
            size_t s = ((size_t)(img * 3) << 20) + ((size_t)gr << 10) + (size_t)gc;
            v[0] = (__bf16)src[s];
            v[1] = (__bf16)src[s + (1u << 20)];
            v[2] = (__bf16)src[s + 2u * (1u << 20)];
        }
        lds4[p] = v;
    }
    if (tid == 0) { v4bf z = {}; lds4[240] = z; }
    __syncthreads();

    const int lane    = tid & 31;
    const int wv      = tid >> 5;
    const int n       = lane & 15;
    const int hi      = lane >> 4;
    const int pixBase = wv * 20 + n;

    v8f acc = {};
#pragma unroll
    for (int kIter = 0; kIter < KITERS; ++kIter) {
        const v16bf a = *(const v16bf*)(aFrag + ((kIter * 32 + lane) << 4));
        v16bf b;
#pragma unroll
        for (int j = 0; j < 4; ++j) {          // 4 taps x 4 channels = 16 halves
            const int tLo = kIter * 8 + j;     // compile-time tap (hi == 0)
            const int tHi = kIter * 8 + 4 + j; // compile-time tap (hi == 1)
            const int iLo = (tLo < 25) ? (pixBase + (tLo / 5) * 20 + (tLo % 5)) : 240;
            const int iHi = (tHi < 25) ? (pixBase + (tHi / 5) * 20 + (tHi % 5)) : 240;
            v4bf pj = lds4[hi ? iHi : iLo];    // one cndmask + ds_load_b64
            b[4 * j + 0] = pj[0];
            b[4 * j + 1] = pj[1];
            b[4 * j + 2] = pj[2];
            b[4 * j + 3] = pj[3];
        }
        acc = __builtin_amdgcn_wmma_f32_16x16x32_bf16(
            false, a, false, b, (short)0, acc, false, false);
    }

    const int gr = tileRow * 8 + wv;
    const int gc = tileCol * 16 + n;
    size_t pix = (((size_t)img << 10) + gr);
    pix = (pix << 10) + gc;
    v8bf o;
#pragma unroll
    for (int r = 0; r < 8; ++r) {
        float v = acc[r] + bias[hi * 8 + r];
        o[r] = (__bf16)(v > 0.0f ? v : 0.0f);
    }
    *(v8bf*)(dst + pix * 16 + hi * 8) = o;
}

// ---------------------------------------------------------------------------
// Filtering: softmax over guide[0:4] (NHWC8 f32), 4-level dilated 3x3 stencil
// with periodic wrap (jnp.roll). Block = 32x8 pixel tile; a 48x24 wrapped
// halo tile of the 3 input channels is staged in LDS as padded 4-float pixels
// so every tap is one aligned ds_load_b128 (36 taps/thread, zero global
// re-reads of neighbors).
// ---------------------------------------------------------------------------
__global__ __launch_bounds__(256) void filtering_kernel(
    const float* __restrict__ imgs, const float* __restrict__ guide,
    float* __restrict__ out)
{
    constexpr int TW = 32, TH = 8, HALO = 8;
    constexpr int LW = TW + 2 * HALO;          // 48
    constexpr int LH = TH + 2 * HALO;          // 24
    __shared__ float ldsF[LH][LW][4];          // padded NHWC4, 18 KiB

    const int img   = blockIdx.z;
    const int tileY = blockIdx.y;              // * TH
    const int tileX = blockIdx.x;              // * TW
    const int tid   = threadIdx.x;
    const size_t plane = (size_t)1 << 20;
    const size_t ibase = (size_t)(img * 3) * plane;

    // stage wrapped halo tile (periodic boundary == jnp.roll semantics)
    for (int p = tid; p < LH * LW; p += 256) {
        int row = p / LW;
        int col = p - row * LW;
        int gy = (tileY * TH + row - HALO) & 1023;
        int gx = (tileX * TW + col - HALO) & 1023;
        size_t s = ibase + ((size_t)gy << 10) + (size_t)gx;
        float4 v;
        v.x = imgs[s];
        v.y = imgs[s + plane];
        v.z = imgs[s + 2 * plane];
        v.w = 0.0f;
        *(float4*)&ldsF[row][col][0] = v;      // ds_store_b128
    }
    __syncthreads();

    const int lx = tid & (TW - 1);
    const int ly = tid >> 5;
    const int px = tileX * TW + lx;
    const int py = tileY * TH + ly;

    const float* g = guide + ((((size_t)img << 10) + py) * 1024 + px) * 8;
    float gv[8];
#pragma unroll
    for (int c = 0; c < 8; ++c) gv[c] = g[c];

    float mx = fmaxf(fmaxf(gv[0], gv[1]), fmaxf(gv[2], gv[3]));
    float e[4], sum = 0.0f;
#pragma unroll
    for (int i = 0; i < 4; ++i) { e[i] = __expf(gv[i] - mx); sum += e[i]; }
    const float inv_sum = 1.0f / sum;

    float acc0 = 0.0f, acc1 = 0.0f, acc2 = 0.0f;
#pragma unroll
    for (int l = 0; l < 4; ++l) {
        const int d = 1 << l;                  // dilation 1,2,4,8 <= HALO
        const float s  = gv[4 + l];
        const float k1 = __expf(-s);           // r2 == 1
        const float k2 = k1 * k1;              // r2 == 2
        const float den = 1.0f + 4.0f * k1 + 4.0f * k2;
        float n0 = 0.0f, n1 = 0.0f, n2 = 0.0f;
#pragma unroll
        for (int di = -1; di <= 1; ++di)
#pragma unroll
            for (int dj = -1; dj <= 1; ++dj) {
                const int r2 = di * di + dj * dj;
                const float kk = (r2 == 0) ? 1.0f : ((r2 == 1) ? k1 : k2);
                const int sy = ly + HALO - di * d;   // roll => src = i - shift
                const int sx = lx + HALO - dj * d;
                float4 p = *(const float4*)&ldsF[sy][sx][0];   // ds_load_b128
                n0 += kk * p.x;
                n1 += kk * p.y;
                n2 += kk * p.z;
            }
        const float wl = e[l] * inv_sum / den;
        acc0 += wl * n0; acc1 += wl * n1; acc2 += wl * n2;
    }
    const size_t o = ibase + ((size_t)py << 10) + (size_t)px;
    out[o]             = acc0;
    out[o + plane]     = acc1;
    out[o + 2 * plane] = acc2;
}

// ---------------------------------------------------------------------------
// Workspace layout (bytes from d_ws):
//   [l*16384)            : packed A-fragments per layer (<= 13.3 KB)
//   [65536 + l*256)      : fused bias per layer (16 floats)
//   [1 MiB, +128 MiB)    : ping plane  (NHWC16 bf16, 4x1024x1024x16)
//   [1 MiB+128 MiB, +128 MiB): pong plane / final NHWC8 f32 guide
// ---------------------------------------------------------------------------
extern "C" void kernel_launch(void* const* d_in, const int* in_sizes, int n_in,
                              void* d_out, int out_size, void* d_ws, size_t ws_size,
                              hipStream_t stream) {
    (void)in_sizes; (void)n_in; (void)out_size; (void)ws_size;

    const float* imgs = (const float*)d_in[0];
    char* ws = (char*)d_ws;

    __bf16* aFrag[4];
    float*  bias[4];
    for (int l = 0; l < 4; ++l) {
        aFrag[l] = (__bf16*)(ws + (size_t)l * 16384);
        bias[l]  = (float*)(ws + 65536 + (size_t)l * 256);
    }
    const size_t PLANE_BYTES = (size_t)4 * 1024 * 1024 * 16 * 2;   // 128 MiB
    __bf16* bufA  = (__bf16*)(ws + (1u << 20));
    __bf16* bufB  = (__bf16*)(ws + (1u << 20) + PLANE_BYTES);
    float*  guide = (float*)bufB;   // layer3 reads bufA, writes here (bufB dead)

    // layer configs: Ci, Cpad, Co, identity
    const int Ci[4]   = {3, 16, 16, 16};
    const int Cpad[4] = {4, 16, 16, 16};
    const int Co[4]   = {16, 16, 16, 8};
    const int idn[4]  = {0, 1, 1, 0};
    for (int l = 0; l < 4; ++l) {
        const int kIters = (25 * Cpad[l] + 31) / 32;
        prep_layer_kernel<<<dim3(1), dim3(256), 0, stream>>>(
            (const float*)d_in[1 + 6 * l + 0], (const float*)d_in[1 + 6 * l + 1],
            (const float*)d_in[1 + 6 * l + 2], (const float*)d_in[1 + 6 * l + 3],
            (const float*)d_in[1 + 6 * l + 4], (const float*)d_in[1 + 6 * l + 5],
            aFrag[l], bias[l], Ci[l], Cpad[l], Co[l], idn[l], kIters);
    }

    dim3 block(256);
    conv_in_wmma<<<dim3(64, 128, 4), block, 0, stream>>>(imgs, bufA, aFrag[0], bias[0]);
    conv16_wmma<16, false><<<dim3(64, 64, 4), block, 0, stream>>>(bufA, bufB, aFrag[1], bias[1]);
    conv16_wmma<16, false><<<dim3(64, 64, 4), block, 0, stream>>>(bufB, bufA, aFrag[2], bias[2]);
    conv16_wmma<8,  true ><<<dim3(64, 64, 4), block, 0, stream>>>(bufA, (void*)guide, aFrag[3], bias[3]);

    filtering_kernel<<<dim3(32, 128, 4), block, 0, stream>>>(imgs, guide, (float*)d_out);
}
